// FuseLayer_38654705664799
// MI455X (gfx1250) — matrix operand order repaired and checked by reading, not compiled, not run on hardware
//
#include <hip/hip_runtime.h>
#include <cstdint>

#define EPS   1e-8f
#define T_LEN 16000
#define INV_T (1.0f / 16000.0f)

typedef __attribute__((ext_vector_type(16))) __bf16          v16bf;
typedef __attribute__((ext_vector_type(8)))  float           v8f;
typedef __attribute__((ext_vector_type(8)))  unsigned short  v8us;

union V16 { v16bf bf; v8us us[2]; };

__device__ __forceinline__ unsigned short f2bf(float f) {
    unsigned u = __builtin_bit_cast(unsigned, f);
    u += 0x7fffu + ((u >> 16) & 1u);           // round-to-nearest-even
    return (unsigned short)(u >> 16);
}

__device__ __forceinline__ v8f wmma_bf16(V16 a, V16 b, v8f c) {
    // D = A(16x32 bf16) * B(32x16 bf16) + C(16x16 f32)
    return __builtin_amdgcn_wmma_f32_16x16x32_bf16(
        /*neg_a=*/false, a.bf, /*neg_b=*/false, b.bf,
        /*c_mod=*/(short)0, c, /*reuse_a=*/false, /*reuse_b=*/false);
}

// ---------------------------------------------------------------- utilities
__global__ void k_zero(float* p, int n) {
    int i = blockIdx.x * blockDim.x + threadIdx.x;
    if (i < n) p[i] = 0.f;
}

// --------------------------------------------- pass 1: per-(b,c) x moments
__global__ void __launch_bounds__(256) k_xstats(const float* __restrict__ x,
                                                float* __restrict__ S1,
                                                float* __restrict__ S2) {
    __shared__ float r1[256], r2[256];
    int bc = blockIdx.x;                                  // b*256 + c
    const float4* row = (const float4*)(x + (size_t)bc * T_LEN);
    float s = 0.f, q = 0.f;
    for (int i = threadIdx.x; i < T_LEN / 4; i += 256) {
        float4 v = row[i];
        s += v.x + v.y + v.z + v.w;
        q += v.x * v.x + v.y * v.y + v.z * v.z + v.w * v.w;
    }
    r1[threadIdx.x] = s; r2[threadIdx.x] = q; __syncthreads();
    for (int st = 128; st > 0; st >>= 1) {
        if (threadIdx.x < st) { r1[threadIdx.x] += r1[threadIdx.x + st];
                                r2[threadIdx.x] += r2[threadIdx.x + st]; }
        __syncthreads();
    }
    if (threadIdx.x == 0) { S1[bc] = r1[0]; S2[bc] = r2[0]; }
}

// ------------------------------------------ prep 1: compose affine chain
__device__ __forceinline__ float blockMean256(float v, float* red) {
    red[threadIdx.x] = v; __syncthreads();
    for (int s = 128; s > 0; s >>= 1) {
        if (threadIdx.x < s) red[threadIdx.x] += red[threadIdx.x + s];
        __syncthreads();
    }
    float m = red[0] * (1.f / 256.f); __syncthreads();
    return m;
}

__device__ __forceinline__ void gn_stage(float& A, float& D, float w, float bb,
                                         float mu, float m2, float* red) {
    float muc = A * mu + D;                                // per-c mean of h
    float m2c = A * A * m2 + 2.f * A * D * mu + D * D;     // per-c E[h^2]
    float M = blockMean256(muc, red);
    float V = blockMean256(m2c, red) - M * M;
    float r = rsqrtf(V + EPS);
    float Ap = r * w, Dp = bb - M * r * w;                 // h' = Ap*h + Dp
    D = Ap * D + Dp; A = Ap * A;
}

__global__ void __launch_bounds__(256) k_prep1(
    const float* __restrict__ S1, const float* __restrict__ S2,
    const float* __restrict__ cue,
    const float* __restrict__ gna_w, const float* __restrict__ gna_b,
    const float* __restrict__ gnn_w, const float* __restrict__ gnn_b,
    const float* __restrict__ f1g_w, const float* __restrict__ f1g_b,
    const float* __restrict__ f1b_w, const float* __restrict__ f1b_b,
    const float* __restrict__ ln1_w, const float* __restrict__ ln1_b,
    const float* __restrict__ cds_w,
    const float* __restrict__ f2g_w, const float* __restrict__ f2g_b,
    const float* __restrict__ f2b_w, const float* __restrict__ f2b_b,
    float* __restrict__ A4, float* __restrict__ D4,
    float* __restrict__ G2, float* __restrict__ B2) {
    __shared__ float cueS[512];
    __shared__ float red[256];
    __shared__ float cue2S[256];
    int b = blockIdx.x, c = threadIdx.x, bc = b * 256 + c;
    cueS[c] = cue[b * 512 + c]; cueS[c + 256] = cue[b * 512 + 256 + c];
    __syncthreads();

    float mu = S1[bc] * INV_T, m2 = S2[bc] * INV_T;
    float A = 1.f, D = 0.f;
    gn_stage(A, D, gna_w[c], gna_b[c], mu, m2, red);       // norm_audio
    gn_stage(A, D, gnn_w[c], gnn_b[c], mu, m2, red);       // norm

    float g = f1g_b[c], bt = f1b_b[c];                     // film1
    for (int k = 0; k < 512; ++k) {
        float e = cueS[k];
        g  += e * f1g_w[c * 512 + k];
        bt += e * f1b_w[c * 512 + k];
    }
    g = fmaxf(g, 0.f) + 1.f; bt = fmaxf(bt, 0.f);          // +residual
    D = g * D + bt; A = g * A;

    gn_stage(A, D, ln1_w[c], ln1_b[c], mu, m2, red);       // layer_norm1
    A4[bc] = A; D4[bc] = D;

    float c2 = 0.f;                                        // cue_down_sample2
    for (int k = 0; k < 512; ++k) c2 += cueS[k] * cds_w[c * 512 + k];
    cue2S[c] = fmaxf(c2, 0.f);
    __syncthreads();
    if (c < 128) {                                         // film2 gamma/beta
        float g2 = f2g_b[c], b2 = f2b_b[c];
        for (int j = 0; j < 256; ++j) {
            float e = cue2S[j];
            g2 += e * f2g_w[c * 256 + j];
            b2 += e * f2b_w[c * 256 + j];
        }
        G2[b * 128 + c] = fmaxf(g2, 0.f) + 1.f;            // store (gamma+1)
        B2[b * 128 + c] = fmaxf(b2, 0.f);
    }
}

// ------------------------------------------- weight fp32 -> bf16 convert
__global__ void k_wconv(const float* __restrict__ w1, const float* __restrict__ w2,
                        unsigned short* __restrict__ w1bf,
                        unsigned short* __restrict__ w2bf) {
    int i = blockIdx.x * 256 + threadIdx.x;
    if (i < 128 * 256) w1bf[i] = f2bf(w1[i]);
    if (i < 256 * 128) w2bf[i] = f2bf(w2[i]);
}

// -------- shared producer: stage normalized x tile into LDS as bf16 [t][c]
#define LDY 272   // 256 + 16 pad, keeps rows 16B aligned
#define LDZ 136   // 128 + 8  pad

__device__ __forceinline__ void produce_y(const float* __restrict__ x, int b, int t0,
                                          const float* sA, const float* sD,
                                          unsigned short* sY) {
    int tid = threadIdx.x;
    for (int it = 0; it < 16; ++it) {
        int c  = it * 16 + (tid >> 4);
        int tl = (tid & 15) * 4;
        const float4 v = *(const float4*)(x + ((size_t)(b * 256 + c)) * T_LEN + t0 + tl);
        float a = sA[c], d = sD[c];
        sY[(tl + 0) * LDY + c] = f2bf(a * v.x + d);
        sY[(tl + 1) * LDY + c] = f2bf(a * v.y + d);
        sY[(tl + 2) * LDY + c] = f2bf(a * v.z + d);
        sY[(tl + 3) * LDY + c] = f2bf(a * v.w + d);
    }
}

// ----------------- pass 2: z = conv1(y) stats only (Σz, Σz² per (b,o))
__global__ void __launch_bounds__(256) k_zstats(
    const float* __restrict__ x, const unsigned short* __restrict__ w1bf,
    const float* __restrict__ A4, const float* __restrict__ D4,
    float* __restrict__ Zs1, float* __restrict__ Zs2) {
    __shared__ unsigned short sY[64 * LDY];
    __shared__ float sA[256], sD[256];
    int b = blockIdx.y, t0 = blockIdx.x * 64, tid = threadIdx.x;
    sA[tid] = A4[b * 256 + tid]; sD[tid] = D4[b * 256 + tid];
    __syncthreads();
    produce_y(x, b, t0, sA, sD, sY);
    __syncthreads();

    int w = tid >> 5, l = tid & 31, hl = l >> 4, ll = l & 15;
    v8f acc[4];
    #pragma unroll
    for (int n = 0; n < 4; ++n) acc[n] = (v8f){0.f,0.f,0.f,0.f,0.f,0.f,0.f,0.f};

    #pragma unroll
    for (int kk = 0; kk < 8; ++kk) {               // K = 256 in steps of 32
        int kb = kk * 32;
        V16 Aop;
        const v8us* pa = (const v8us*)(w1bf + (16 * w + ll) * 256 + kb + 8 * hl);
        Aop.us[0] = pa[0]; Aop.us[1] = pa[2];      // +16 elems = +2 v8us
        #pragma unroll
        for (int n = 0; n < 4; ++n) {
            V16 Bop;
            const unsigned short* pb = &sY[(16 * n + ll) * LDY + kb + 8 * hl];
            Bop.us[0] = *(const v8us*)pb; Bop.us[1] = *(const v8us*)(pb + 16);
            acc[n] = wmma_bf16(Aop, Bop, acc[n]);
        }
    }
    #pragma unroll
    for (int r = 0; r < 8; ++r) {
        float p = 0.f, q = 0.f;
        #pragma unroll
        for (int n = 0; n < 4; ++n) { float v = acc[n][r]; p += v; q += v * v; }
        #pragma unroll
        for (int m = 1; m < 16; m <<= 1) { p += __shfl_xor(p, m, 32);
                                           q += __shfl_xor(q, m, 32); }
        if (ll == 0) {
            int o = 16 * w + r + 8 * hl;
            atomicAdd(&Zs1[b * 128 + o], p);
            atomicAdd(&Zs2[b * 128 + o], q);
        }
    }
}

// ----------------------------- prep 2: compose film2+ln2 affine (A5, D5)
__global__ void __launch_bounds__(128) k_prep2(
    const float* __restrict__ Zs1, const float* __restrict__ Zs2,
    const float* __restrict__ G2, const float* __restrict__ B2,
    const float* __restrict__ ln2_w, const float* __restrict__ ln2_b,
    float* __restrict__ A5, float* __restrict__ D5) {
    __shared__ float red[128];
    int b = blockIdx.x, o = threadIdx.x, bo = b * 128 + o;
    float muz = Zs1[bo] * INV_T, m2z = Zs2[bo] * INV_T;
    float a = G2[bo], d = B2[bo];                          // z2 = a*z + d
    float mu2 = a * muz + d;
    float m22 = a * a * m2z + 2.f * a * d * muz + d * d;
    red[o] = mu2; __syncthreads();
    for (int s = 64; s > 0; s >>= 1) { if (o < s) red[o] += red[o + s]; __syncthreads(); }
    float M = red[0] * (1.f / 128.f); __syncthreads();
    red[o] = m22; __syncthreads();
    for (int s = 64; s > 0; s >>= 1) { if (o < s) red[o] += red[o + s]; __syncthreads(); }
    float V = red[0] * (1.f / 128.f) - M * M;
    float r = rsqrtf(V + EPS);
    A5[bo] = ln2_w[o] * r * a;
    D5[bo] = ln2_w[o] * r * (d - M) + ln2_b[o];
}

// ------------ pass 3: fused  y -> conv1 -> affine -> conv2 -> out write
__global__ void __launch_bounds__(256) k_fused(
    const float* __restrict__ x,
    const unsigned short* __restrict__ w1bf, const unsigned short* __restrict__ w2bf,
    const float* __restrict__ A4, const float* __restrict__ D4,
    const float* __restrict__ A5, const float* __restrict__ D5,
    float* __restrict__ out) {
    __shared__ unsigned short sY[64 * LDY];
    __shared__ unsigned short sZ[64 * LDZ];
    __shared__ float sA[256], sD[256], sA5[128], sD5[128];
    int b = blockIdx.y, t0 = blockIdx.x * 64, tid = threadIdx.x;
    sA[tid] = A4[b * 256 + tid]; sD[tid] = D4[b * 256 + tid];
    if (tid < 128) { sA5[tid] = A5[b * 128 + tid]; sD5[tid] = D5[b * 128 + tid]; }
    __syncthreads();
    produce_y(x, b, t0, sA, sD, sY);
    __syncthreads();

    int w = tid >> 5, l = tid & 31, hl = l >> 4, ll = l & 15;

    // ---- GEMM1: z[128 x 64] = w1[128x256] * y[256x64]
    v8f acc1[4];
    #pragma unroll
    for (int n = 0; n < 4; ++n) acc1[n] = (v8f){0.f,0.f,0.f,0.f,0.f,0.f,0.f,0.f};
    #pragma unroll
    for (int kk = 0; kk < 8; ++kk) {
        int kb = kk * 32;
        V16 Aop;
        const v8us* pa = (const v8us*)(w1bf + (16 * w + ll) * 256 + kb + 8 * hl);
        Aop.us[0] = pa[0]; Aop.us[1] = pa[2];
        #pragma unroll
        for (int n = 0; n < 4; ++n) {
            V16 Bop;
            const unsigned short* pb = &sY[(16 * n + ll) * LDY + kb + 8 * hl];
            Bop.us[0] = *(const v8us*)pb; Bop.us[1] = *(const v8us*)(pb + 16);
            acc1[n] = wmma_bf16(Aop, Bop, acc1[n]);
        }
    }
    // apply film2+ln2 affine, stash as bf16 in LDS [t][o]
    #pragma unroll
    for (int n = 0; n < 4; ++n) {
        #pragma unroll
        for (int r = 0; r < 8; ++r) {
            int o = 16 * w + r + 8 * hl;
            float h = sA5[o] * acc1[n][r] + sD5[o];
            sZ[(16 * n + ll) * LDZ + o] = f2bf(h);
        }
    }
    __syncthreads();

    // ---- GEMM2: out[256 x 64] = w2[256x128] * z2[128x64]
    v8f acc2[2][4];
    #pragma unroll
    for (int g = 0; g < 2; ++g)
        #pragma unroll
        for (int n = 0; n < 4; ++n)
            acc2[g][n] = (v8f){0.f,0.f,0.f,0.f,0.f,0.f,0.f,0.f};
    #pragma unroll
    for (int kk = 0; kk < 4; ++kk) {               // K = 128 in steps of 32
        int kb = kk * 32;
        #pragma unroll
        for (int g = 0; g < 2; ++g) {
            V16 Aop;
            const v8us* pa = (const v8us*)(w2bf + (16 * w + 128 * g + ll) * 128 + kb + 8 * hl);
            Aop.us[0] = pa[0]; Aop.us[1] = pa[2];
            #pragma unroll
            for (int n = 0; n < 4; ++n) {
                V16 Bop;
                const unsigned short* pb = &sZ[(16 * n + ll) * LDZ + kb + 8 * hl];
                Bop.us[0] = *(const v8us*)pb; Bop.us[1] = *(const v8us*)(pb + 16);
                acc2[g][n] = wmma_bf16(Aop, Bop, acc2[g][n]);
            }
        }
    }
    #pragma unroll
    for (int g = 0; g < 2; ++g)
        #pragma unroll
        for (int n = 0; n < 4; ++n)
            #pragma unroll
            for (int r = 0; r < 8; ++r) {
                int q = 16 * w + 128 * g + r + 8 * hl;
                out[((size_t)(b * 256 + q)) * T_LEN + t0 + 16 * n + ll] = acc2[g][n][r];
            }
}

// ---------------------------------------------------------------- launcher
extern "C" void kernel_launch(void* const* d_in, const int* in_sizes, int n_in,
                              void* d_out, int out_size, void* d_ws, size_t ws_size,
                              hipStream_t stream) {
    (void)in_sizes; (void)n_in; (void)out_size; (void)ws_size;
    const float* x      = (const float*)d_in[0];
    const float* cue    = (const float*)d_in[1];
    const float* gna_w  = (const float*)d_in[2];
    const float* gna_b  = (const float*)d_in[3];
    const float* gnn_w  = (const float*)d_in[4];
    const float* gnn_b  = (const float*)d_in[5];
    const float* f1g_w  = (const float*)d_in[6];
    const float* f1g_b  = (const float*)d_in[7];
    const float* f1b_w  = (const float*)d_in[8];
    const float* f1b_b  = (const float*)d_in[9];
    const float* ln1_w  = (const float*)d_in[10];
    const float* ln1_b  = (const float*)d_in[11];
    const float* conv1w = (const float*)d_in[12];
    const float* cds_w  = (const float*)d_in[13];
    const float* f2g_w  = (const float*)d_in[14];
    const float* f2g_b  = (const float*)d_in[15];
    const float* f2b_w  = (const float*)d_in[16];
    const float* f2b_b  = (const float*)d_in[17];
    const float* ln2_w  = (const float*)d_in[18];
    const float* ln2_b  = (const float*)d_in[19];
    const float* conv2w = (const float*)d_in[20];

    char* ws = (char*)d_ws;                        // ~240 KB total
    float* S1  = (float*)(ws + 0);                 // [16,256]
    float* S2  = (float*)(ws + 16384);             // [16,256]
    float* A4  = (float*)(ws + 32768);             // [16,256]
    float* D4  = (float*)(ws + 49152);             // [16,256]
    float* G2  = (float*)(ws + 65536);             // [16,128]
    float* B2  = (float*)(ws + 73728);             // [16,128]
    float* Zs1 = (float*)(ws + 81920);             // [16,128] (atomic acc)
    float* Zs2 = (float*)(ws + 90112);             // [16,128] (atomic acc)
    float* A5  = (float*)(ws + 98304);             // [16,128]
    float* D5  = (float*)(ws + 106496);            // [16,128]
    unsigned short* w1bf = (unsigned short*)(ws + 114688); // [128,256] bf16
    unsigned short* w2bf = (unsigned short*)(ws + 180224); // [256,128] bf16

    k_zero  <<<dim3(16),       256, 0, stream>>>(Zs1, 4096);   // Zs1+Zs2
    k_xstats<<<dim3(4096),     256, 0, stream>>>(x, S1, S2);
    k_prep1 <<<dim3(16),       256, 0, stream>>>(S1, S2, cue, gna_w, gna_b,
                 gnn_w, gnn_b, f1g_w, f1g_b, f1b_w, f1b_b, ln1_w, ln1_b,
                 cds_w, f2g_w, f2g_b, f2b_w, f2b_b, A4, D4, G2, B2);
    k_wconv <<<dim3(128),      256, 0, stream>>>(conv1w, conv2w, w1bf, w2bf);
    k_zstats<<<dim3(250, 16),  256, 0, stream>>>(x, w1bf, A4, D4, Zs1, Zs2);
    k_prep2 <<<dim3(16),       128, 0, stream>>>(Zs1, Zs2, G2, B2, ln2_w, ln2_b, A5, D5);
    k_fused <<<dim3(250, 16),  256, 0, stream>>>(x, w1bf, w2bf, A4, D4, A5, D5,
                                                 (float*)d_out);
}
